// BallNCL_42571715838006
// MI455X (gfx1250) — matrix operand order repaired and checked by reading, not compiled, not run on hardware
//
#include <hip/hip_runtime.h>

typedef __attribute__((ext_vector_type(16))) __bf16 v16bf;
typedef __attribute__((ext_vector_type(8)))  float  v8f;

#define HDIM 512
#define NPTS 4            // points per workgroup
#define NCOL (NPTS * 16)  // 64 padded channel columns
#define NTHREADS 256
#define WELEMS (HDIM * HDIM) // 262144 elements per weight matrix

// LDS padding to avoid bank conflicts:
//  - activation columns: stride 528 bf16 (1056 B -> 264 words == 8 mod 64)
//  - preactivation rows: stride 68 f32  (68 words == 4 mod 64)
#define HSTRIDE 528
#define ASTRIDE 68
#define HBBYTES (NCOL * HSTRIDE * 2)          // 67584 B per activation buffer
#define APREBYTES (HDIM * ASTRIDE * 4)        // 139264 B
#define SMEM_TOTAL (2 * HBBYTES + APREBYTES)  // 274432 B

// Channel layout per point (16 columns, col 15 = zero pad):
//   0      : value
//   1..4   : d/dz_k
//   5..14  : d2/dz_j dz_k for pairs (j<=k) in order
//            (0,0)(0,1)(0,2)(0,3)(1,1)(1,2)(1,3)(2,2)(2,3)(3,3)
// PAIRD2[j][i] = index (0..9) of unordered pair {j,i} in that order.
__constant__ int PAIRD2[4][4] = {
    {0, 1, 2, 3}, {1, 4, 5, 6}, {2, 5, 7, 8}, {3, 6, 8, 9}};

// softplus(25a)/25 and its first/second derivatives
__device__ __forceinline__ void act_eval(float a, float& h, float& s1, float& s2) {
  float t = 25.f * a;
  float s = 1.f / (1.f + __expf(-t)); // sigmoid(25a), overflow-safe
  s1 = s;
  s2 = 25.f * s * (1.f - s);
  h = (t > 15.f) ? a : 0.04f * __logf(1.f + __expf(t));
}

// ---------------------------------------------------------------------------
// Prep: convert W1/W2 (f32 row-major 512x512) to bf16, pre-swizzled into the
// contiguous-k (k = 16*(lane>>4) + s) WMMA A-fragment order so each fragment is ONE coalesced
// 32-byte load per lane:
//   dst element index = (((mt*16 + ks)*32 + lane)*16 + e)
//   src: row r = mt*16 + (lane&15), col kb = ks*32 + ((lane>>4)<<4) + hf*8 + e
// ---------------------------------------------------------------------------
__global__ void prep_swizzle_kernel(const float* __restrict__ W1,
                                    const float* __restrict__ W2,
                                    __bf16* __restrict__ ws) {
  int j = blockIdx.x * NTHREADS + threadIdx.x; // 0 .. 2*WELEMS-1
  const float* src = (j < WELEMS) ? W1 : W2;
  __bf16* dst = ws + ((j < WELEMS) ? 0 : WELEMS);
  int i = j & (WELEMS - 1);
  int e = i & 7;
  int c8 = i >> 3;              // 8-bf16 chunk id
  int hf = c8 & 1;              // low/high 16-byte half of fragment
  int lane = (c8 >> 1) & 31;
  int ks = (c8 >> 6) & 15;
  int mt = c8 >> 10;
  int r = mt * 16 + (lane & 15);
  int kb = ks * 32 + ((lane >> 4) << 4) + hf * 8 + e;
  dst[i] = (__bf16)src[r * HDIM + kb];
}

// ---------------------------------------------------------------------------
// One 512x512 layer GEMM: Apre[m][n] = sum_k W[m,k] * Hin[n][k]
//   Ws  : bf16 weights, pre-swizzled A-fragment layout (global, L2-resident)
//   Hin : LDS, [NCOL][HSTRIDE] bf16 (K contiguous)  -> B fragments
//   Apre: LDS, [HDIM][ASTRIDE] f32
// Each wave owns 4 m-tiles x 4 n-tiles; k-outer loop keeps all 16
// accumulator tiles live so each B fragment is read from LDS exactly once.
// ---------------------------------------------------------------------------
__device__ __forceinline__ void layer_gemm(const __bf16* __restrict__ Ws,
                                           const __bf16* Hin, float* Apre,
                                           int wave, int lane) {
  const int nl = lane & 15;
  const int hi = lane >> 4;
  const __bf16* hbase = Hin + nl * HSTRIDE + (hi << 4); // lanes 16..31: K+16
  const __bf16* wbase = Ws + (wave * 4 * 16 * 32 + lane) * 16;

  v8f acc[4][4];
#pragma unroll
  for (int mi = 0; mi < 4; ++mi)
#pragma unroll
    for (int nt = 0; nt < 4; ++nt) acc[mi][nt] = (v8f){};

  for (int ks = 0; ks < 16; ++ks) {
    v16bf bfr[4];
#pragma unroll
    for (int nt = 0; nt < 4; ++nt)
      bfr[nt] = *(const v16bf*)(hbase + ks * 32 + nt * 16 * HSTRIDE);
#pragma unroll
    for (int mi = 0; mi < 4; ++mi) {
      v16bf af = *(const v16bf*)(wbase + (mi * 16 + ks) * (32 * 16));
#pragma unroll
      for (int nt = 0; nt < 4; ++nt)
        acc[mi][nt] = __builtin_amdgcn_wmma_f32_16x16x32_bf16(
            false, af, false, bfr[nt], (short)0, acc[mi][nt], false, false);
    }
  }

  // C/D layout: lane gives N (=lane&15); VGPR v gives M = v + 8*(lane>>4)
#pragma unroll
  for (int mi = 0; mi < 4; ++mi) {
    const int mb = (wave * 4 + mi) * 16 + (hi << 3);
#pragma unroll
    for (int v = 0; v < 8; ++v) {
      float* row = Apre + (mb + v) * ASTRIDE + nl;
      row[0] = acc[mi][0][v];
      row[16] = acc[mi][1][v];
      row[32] = acc[mi][2][v];
      row[48] = acc[mi][3][v];
    }
  }
}

// ---------------------------------------------------------------------------
// Row-local jet nonlinearity: consumes pre-activation channels (f32, +bias on
// the value channel only), emits next-layer channels in bf16 [NCOL][HSTRIDE].
//   h      = sigma(a)
//   dh_k   = s1 * da_k
//   d2h_jk = s2 * da_j * da_k + s1 * d2a_jk
// ---------------------------------------------------------------------------
__device__ __forceinline__ void layer_pointwise(const float* Apre,
                                                const float* __restrict__ bias,
                                                __bf16* Hout, int tid) {
  for (int it = 0; it < 8; ++it) {
    int q = tid + NTHREADS * it; // 0..2047 = (m, p)
    int p = q & 3, m = q >> 2;
    const float* ap = Apre + m * ASTRIDE + p * 16;
    float av[15];
#pragma unroll
    for (int c = 0; c < 15; ++c) av[c] = ap[c];
    float h, s1, s2;
    act_eval(av[0] + bias[m], h, s1, s2);
    __bf16* col = Hout + p * 16 * HSTRIDE + m;
    col[0] = (__bf16)h;
    float d1[4];
#pragma unroll
    for (int k = 0; k < 4; ++k) {
      d1[k] = av[1 + k];
      col[(1 + k) * HSTRIDE] = (__bf16)(s1 * d1[k]);
    }
    int ci = 5;
#pragma unroll
    for (int j = 0; j < 4; ++j)
#pragma unroll
      for (int k = j; k < 4; ++k) {
        col[ci * HSTRIDE] = (__bf16)(s2 * d1[j] * d1[k] + s1 * av[ci]);
        ++ci;
      }
    col[15 * HSTRIDE] = (__bf16)0.f; // keep pad column zero
  }
}

// ---------------------------------------------------------------------------
// Main kernel: one workgroup = 4 points, full pipeline in LDS.
// ---------------------------------------------------------------------------
__global__ void __launch_bounds__(NTHREADS)
mlp_trace_kernel(const float* __restrict__ x, const float* __restrict__ W0,
                 const float* __restrict__ b0, const __bf16* __restrict__ W1s,
                 const float* __restrict__ b1, const __bf16* __restrict__ W2s,
                 const float* __restrict__ b2, const float* __restrict__ W3,
                 const float* __restrict__ b3, float* __restrict__ out) {
  extern __shared__ __align__(32) char smem[];
  __bf16* HB0 = (__bf16*)smem;                // [64][528] bf16
  __bf16* HB1 = (__bf16*)(smem + HBBYTES);    // [64][528] bf16
  float* APRE = (float*)(smem + 2 * HBBYTES); // [512][68] f32

  const int tid = (int)threadIdx.x;
  const int lane = tid & 31, wave = tid >> 5;
  const int pt0 = (int)blockIdx.x * NPTS;

  // ---- layer 0 (512x4): seed the jet.  da0_k = W0[:,k], d2a0 = 0.
  for (int it = 0; it < 8; ++it) {
    int q = tid + NTHREADS * it;
    int p = q & 3, m = q >> 2;
    const float* zr = x + (pt0 + p) * 4;
    float w[4], z[4];
#pragma unroll
    for (int k = 0; k < 4; ++k) {
      w[k] = W0[m * 4 + k];
      z[k] = zr[k];
    }
    float a = b0[m];
#pragma unroll
    for (int k = 0; k < 4; ++k) a += w[k] * z[k];
    float h, s1, s2;
    act_eval(a, h, s1, s2);
    __bf16* col = HB0 + p * 16 * HSTRIDE + m;
    col[0] = (__bf16)h;
#pragma unroll
    for (int k = 0; k < 4; ++k) col[(1 + k) * HSTRIDE] = (__bf16)(s1 * w[k]);
    int ci = 5;
#pragma unroll
    for (int j = 0; j < 4; ++j)
#pragma unroll
      for (int k = j; k < 4; ++k) {
        col[ci * HSTRIDE] = (__bf16)(s2 * w[j] * w[k]);
        ++ci;
      }
    col[15 * HSTRIDE] = (__bf16)0.f;
  }
  __syncthreads();

  // ---- layer 1: WMMA GEMM + jet nonlinearity
  layer_gemm(W1s, HB0, APRE, wave, lane);
  __syncthreads();
  layer_pointwise(APRE, b1, HB1, tid);
  __syncthreads();

  // ---- layer 2
  layer_gemm(W2s, HB1, APRE, wave, lane);
  __syncthreads();
  layer_pointwise(APRE, b2, HB0, tid);
  __syncthreads();

  // ---- final linear layer + contractions:
  //   Q[j][i] = W3[j,:] . d2h2_{pair(j,i)}      (16 dots)
  //   P[i]    = W3[i,:] . (sum_j d2h2_{jj})     (4 dots)
  //   u4part  = W3[4,:] . h2                    (1 dot)
  //   trace_i = P[i] - sum_j Q[j][i];  out4 = u4part + b3[4]
  {
    int p = tid & 3, slice = tid >> 2; // 64 slices of 8 rows each
    const __bf16* col = HB0 + p * 16 * HSTRIDE;
    float acc[21];
#pragma unroll
    for (int c = 0; c < 21; ++c) acc[c] = 0.f;
    for (int mm = 0; mm < 8; ++mm) {
      int m = slice * 8 + mm;
      float w3[5];
#pragma unroll
      for (int j = 0; j < 5; ++j) w3[j] = W3[j * HDIM + m];
      float d2[10];
#pragma unroll
      for (int c = 0; c < 10; ++c) d2[c] = (float)col[(5 + c) * HSTRIDE + m];
      float hval = (float)col[m];
      float S = d2[0] + d2[4] + d2[7] + d2[9]; // diagonal pairs (j,j)
#pragma unroll
      for (int j = 0; j < 4; ++j)
#pragma unroll
        for (int i = 0; i < 4; ++i) acc[j * 4 + i] += w3[j] * d2[PAIRD2[j][i]];
#pragma unroll
      for (int i = 0; i < 4; ++i) acc[16 + i] += w3[i] * S;
      acc[20] += w3[4] * hval;
    }
    float* parts = APRE; // reuse scratch: [4][64][21] partials
#pragma unroll
    for (int c = 0; c < 21; ++c) parts[(p * 64 + slice) * 21 + c] = acc[c];
  }
  __syncthreads();
  float* red = APRE + 4 * 64 * 21;
  if (tid < 84) {
    int p = tid / 21, c = tid % 21;
    float s = 0.f;
    for (int sl = 0; sl < 64; ++sl) s += APRE[(p * 64 + sl) * 21 + c];
    red[p * 21 + c] = s;
  }
  __syncthreads();
  if (tid < NPTS * 5) {
    int p = tid / 5, o = tid % 5;
    const float* R = red + p * 21;
    float val;
    if (o < 4)
      val = R[16 + o] - (R[0 + o] + R[4 + o] + R[8 + o] + R[12 + o]);
    else
      val = R[20] + b3[4];
    out[(pt0 + p) * 5 + o] = val;
  }
}

extern "C" void kernel_launch(void* const* d_in, const int* in_sizes, int n_in,
                              void* d_out, int out_size, void* d_ws,
                              size_t ws_size, hipStream_t stream) {
  (void)n_in; (void)out_size; (void)ws_size;
  const float* x  = (const float*)d_in[0];
  const float* W0 = (const float*)d_in[1];
  const float* b0 = (const float*)d_in[2];
  const float* W1 = (const float*)d_in[3];
  const float* b1 = (const float*)d_in[4];
  const float* W2 = (const float*)d_in[5];
  const float* b2 = (const float*)d_in[6];
  const float* W3 = (const float*)d_in[7];
  const float* b3 = (const float*)d_in[8];
  float* out = (float*)d_out;
  __bf16* ws = (__bf16*)d_ws; // [2][262144] bf16 swizzled weights = 1 MB

  // 1) convert + swizzle W1/W2 into WMMA A-fragment order (every call; cheap)
  prep_swizzle_kernel<<<(2 * WELEMS) / NTHREADS, NTHREADS, 0, stream>>>(W1, W2, ws);

  // 2) fused second-order-jet MLP; 274432 B dynamic LDS per workgroup
  (void)hipFuncSetAttribute((const void*)mlp_trace_kernel,
                            hipFuncAttributeMaxDynamicSharedMemorySize,
                            SMEM_TOTAL);
  const int B = in_sizes[0] / 4; // 16384 points
  mlp_trace_kernel<<<B / NPTS, NTHREADS, SMEM_TOTAL, stream>>>(
      x, W0, b0, ws, b1, ws + WELEMS, b2, W3, b3, out);
}
